// FlaggedDimeNetPPEncoder_86354612453475
// MI455X (gfx1250) — matrix operand order repaired and checked by reading, not compiled, not run on hardware
//
#include <hip/hip_runtime.h>
#include <utility>

// ---------------------------------------------------------------------------
// DimeNet++ forward for MI455X (gfx1250, wave32).
//  - All dense 128-wide layers: bf16 WMMA (v_wmma_f32_16x16x32_bf16), f32 accum.
//  - lin_sbf1@lin_sbf2 and lin_rbf1@lin_rbf2 fused into single weights (no
//    nonlinearity between them in the reference).
//  - sbf GEMM fuses the triplet gather (idx_kj) * multiply * scatter (idx_ji)
//    into its epilogue with global_atomic_add_f32 -> no [T,128] intermediate.
//  - Weights pre-swizzled once per launch into WMMA-B-fragment order so each
//    lane loads its 16 bf16 B elements with one 32B load.
// ---------------------------------------------------------------------------

typedef __attribute__((ext_vector_type(16))) __bf16 v16bf;
typedef __attribute__((ext_vector_type(8)))  float  v8f;

#define PI_F 3.14159265358979323846f
#define CUTOFF_INV 0.2f   // 1/5.0

__device__ __forceinline__ float silu_f(float x) { return x / (1.0f + __expf(-x)); }

__device__ __forceinline__ unsigned short f2bf(float f) {
  union { float f; unsigned u; } a; a.f = f;
  unsigned u = a.u;
  u += 0x7FFFu + ((u >> 16) & 1u);   // round-to-nearest-even
  return (unsigned short)(u >> 16);
}

__device__ __forceinline__ float envelope6(float x) {
  // p = PEXP+1 = 6:  1/x - 28 x^5 + 48 x^6 - 21 x^7   (x < 1, else 0)
  if (x >= 1.0f) return 0.0f;
  float x2 = x * x, x4 = x2 * x2, x5 = x4 * x;
  return 1.0f / x + x5 * (-28.0f + x * (48.0f - 21.0f * x));
}

__device__ __forceinline__ void atomAddF(float* p, float v) {
  __hip_atomic_fetch_add(p, v, __ATOMIC_RELAXED, __HIP_MEMORY_SCOPE_AGENT);
}

union AFrag { v16bf v; unsigned short s[16]; };
union CFrag { v8f   v; float          f[8];  };

// A-fragment k index for 16-bit A 16x32 (contiguous: k = 16*hi + e, matches packed B), element e in 0..15.
__device__ __forceinline__ int a_k_of(int e, int hi) {
  int v = e >> 1, h = e & 1;
  (void)v; (void)h;
  return (hi << 4) + e;
}

// ---------------------------------------------------------------------------
// Weight pre-swizzle: W f32 [Kreal,128] (rows k<Kreal; k>=Kreal zero-padded to
// Kpad) -> bf16 blob laid out [kstep][ntile][lane][16] matching the B-matrix
// 32x16 bf16 fragment in this file's contiguous k order (as a_k_of): lane<16 -> K = 2v+h, lane>=16 -> K = 16+2v+h,
// N = ntile*16 + (lane&15).
// ---------------------------------------------------------------------------
__global__ void k_pack_w(const float* __restrict__ W, unsigned short* __restrict__ out,
                         int Kreal, int Kpad) {
  int idx = blockIdx.x * 256 + threadIdx.x;
  int total = Kpad * 128;
  if (idx >= total) return;
  int frag = idx >> 9;            // /512
  int rem  = idx & 511;
  int lane = rem >> 4;
  int e    = rem & 15;
  int v    = e >> 1, h = e & 1;
  int ks   = frag >> 3, nt = frag & 7;
  int k    = (ks << 5) + ((lane >= 16) ? 16 : 0) + (v << 1) + h;
  int n    = (nt << 4) + (lane & 15);
  float val = (k < Kreal) ? W[(long)k * 128 + n] : 0.0f;
  out[idx] = f2bf(val);
}

// Wf[k,n] = sum_j W1[k,j] * W2[j,n]   (small: Kin*128 outputs, 128-deep dot)
__global__ void k_fuse_w(const float* __restrict__ W1, const float* __restrict__ W2,
                         float* __restrict__ Wf, int Kin) {
  int idx = blockIdx.x * 256 + threadIdx.x;
  if (idx >= Kin * 128) return;
  int k = idx >> 7, n = idx & 127;
  float s = 0.0f;
  #pragma unroll 8
  for (int j = 0; j < 128; ++j) s += W1[k * 128 + j] * W2[j * 128 + n];
  Wf[idx] = s;
}

// ---------------------------------------------------------------------------
// Main WMMA GEMM: Y[M,128] = op( X[M,K] @ W[K,128] )  (K multiple of 32)
// FLAGS: 1=bias, 2=silu, 4=accumulate into Y
// 128 threads = 4 waves; wave w computes rows [blk*64 + 16w, +16) x 128 cols.
// ---------------------------------------------------------------------------
template <int FLAGS>
__launch_bounds__(128)
__global__ void k_gemm(const float* __restrict__ X, const unsigned short* __restrict__ Wp,
                       const float* __restrict__ bias, float* __restrict__ Y,
                       int M, int K) {
  __shared__ float lds[64 * 32];
  const int tid = threadIdx.x;
  const int wave = tid >> 5, lane = tid & 31;
  const int m = lane & 15, hi = lane >> 4;
  const long row0 = (long)blockIdx.x * 64;

  CFrag acc[8];
  #pragma unroll
  for (int i = 0; i < 8; ++i)
    #pragma unroll
    for (int j = 0; j < 8; ++j) acc[i].f[j] = 0.0f;

  const int nks = K >> 5;
  for (int ks = 0; ks < nks; ++ks) {
    __syncthreads();
    #pragma unroll
    for (int i = 0; i < 4; ++i) {            // stage 64x32 f32 tile, float4 loads
      int t = tid + i * 128;                 // 0..511
      int r = t >> 3;
      int c = (t & 7) << 2;
      long row = row0 + r;
      float4 val = make_float4(0.f, 0.f, 0.f, 0.f);
      if (row < M) val = *(const float4*)(X + row * (long)K + ks * 32 + c);
      *(float4*)(&lds[r * 32 + c]) = val;
    }
    __syncthreads();

    AFrag a;
    #pragma unroll
    for (int e = 0; e < 16; ++e)
      a.s[e] = f2bf(lds[(wave * 16 + m) * 32 + a_k_of(e, hi)]);

    const unsigned short* wrow = Wp + (long)ks * 8 * 512;
    #pragma unroll
    for (int nt = 0; nt < 8; ++nt) {
      v16bf b = *(const v16bf*)(wrow + nt * 512 + lane * 16);
      acc[nt].v = __builtin_amdgcn_wmma_f32_16x16x32_bf16(
          false, a.v, false, b, (short)0, acc[nt].v, false, false);
    }
  }

  // Epilogue. C/D layout: vgpr r, lane l -> row = r + 8*(l>=16), col = l&15.
  #pragma unroll
  for (int nt = 0; nt < 8; ++nt) {
    int col = nt * 16 + (lane & 15);
    float bv = (FLAGS & 1) ? bias[col] : 0.0f;
    #pragma unroll
    for (int r = 0; r < 8; ++r) {
      long row = row0 + wave * 16 + hi * 8 + r;
      if (row < M) {
        float vv = acc[nt].f[r] + bv;
        if (FLAGS & 2) vv = silu_f(vv);
        float* dst = Y + row * 128 + col;
        if (FLAGS & 4) *dst += vv; else *dst = vv;
      }
    }
  }
}

// ---------------------------------------------------------------------------
// Fused sbf path:  S = sbf[T,64]bf16 @ Wsbf[64,128];  for elem (t,c):
//   atomicAdd(agg[idx_ji[t],c], S(t,c) * xdown[idx_kj[t],c])
// Grid must be exactly T/64 blocks (T multiple of 64).
// ---------------------------------------------------------------------------
__launch_bounds__(128)
__global__ void k_sbf_gemm_scatter(const unsigned short* __restrict__ Sbf,
                                   const unsigned short* __restrict__ Wp,
                                   const float* __restrict__ xdown,
                                   const int* __restrict__ idx_kj,
                                   const int* __restrict__ idx_ji,
                                   float* __restrict__ agg, int T) {
  __shared__ unsigned short lds[64 * 32];
  const int tid = threadIdx.x;
  const int wave = tid >> 5, lane = tid & 31;
  const int m = lane & 15, hi = lane >> 4;
  const long row0 = (long)blockIdx.x * 64;

  CFrag acc[8];
  #pragma unroll
  for (int i = 0; i < 8; ++i)
    #pragma unroll
    for (int j = 0; j < 8; ++j) acc[i].f[j] = 0.0f;

  for (int ks = 0; ks < 2; ++ks) {
    __syncthreads();
    {                                         // stage 64x32 bf16: 32B per thread
      int r = tid >> 1;
      int c = (tid & 1) << 4;
      *(v16bf*)(&lds[r * 32 + c]) =
          *(const v16bf*)(Sbf + (row0 + r) * 64 + ks * 32 + c);
    }
    __syncthreads();

    AFrag a;
    #pragma unroll
    for (int e = 0; e < 16; ++e)
      a.s[e] = lds[(wave * 16 + m) * 32 + a_k_of(e, hi)];

    const unsigned short* wrow = Wp + (long)ks * 8 * 512;
    #pragma unroll
    for (int nt = 0; nt < 8; ++nt) {
      v16bf b = *(const v16bf*)(wrow + nt * 512 + lane * 16);
      acc[nt].v = __builtin_amdgcn_wmma_f32_16x16x32_bf16(
          false, a.v, false, b, (short)0, acc[nt].v, false, false);
    }
  }

  const int col0 = lane & 15;
  for (int r = 0; r < 8; ++r) {
    long t = row0 + wave * 16 + hi * 8 + r;
    if (t >= T) continue;
    const int kj = idx_kj[t], ji = idx_ji[t];
    const float* xrow = xdown + (long)kj * 128;
    float*       arow = agg   + (long)ji * 128;
    #pragma unroll
    for (int nt = 0; nt < 8; ++nt) {
      int col = nt * 16 + col0;
      atomAddF(&arow[col], acc[nt].f[r] * xrow[col]);
    }
  }
}

// ------------------------------ elementwise --------------------------------
__global__ void k_zero(float* __restrict__ y, long long n) {
  long long i = (long long)blockIdx.x * 256 + threadIdx.x;
  if (i < n) y[i] = 0.0f;
}
__global__ void k_add(float* __restrict__ y, const float* __restrict__ a, long long n) {
  long long i = (long long)blockIdx.x * 256 + threadIdx.x;
  if (i < n) y[i] += a[i];
}
__global__ void k_embed(const int* __restrict__ z, const int* __restrict__ tg,
                        const float* __restrict__ emb, const float* __restrict__ tag,
                        float* __restrict__ h, int N) {
  int idx = blockIdx.x * 256 + threadIdx.x;
  if (idx >= N * 128) return;
  int n = idx >> 7, c = idx & 127;
  h[idx] = emb[z[n] * 128 + c] + tag[tg[n] * 128 + c];
}
__global__ void k_edge_geom(const float* __restrict__ pos, const int* __restrict__ ei,
                            const int* __restrict__ ej, float* __restrict__ rel,
                            float* __restrict__ dist, float* __restrict__ rbf, int E) {
  int e = blockIdx.x * 256 + threadIdx.x;
  if (e >= E) return;
  int i = ei[e], j = ej[e];
  float rx = pos[i * 3 + 0] - pos[j * 3 + 0];
  float ry = pos[i * 3 + 1] - pos[j * 3 + 1];
  float rz = pos[i * 3 + 2] - pos[j * 3 + 2];
  rel[e * 3 + 0] = rx; rel[e * 3 + 1] = ry; rel[e * 3 + 2] = rz;
  float d = sqrtf(rx * rx + ry * ry + rz * rz + 1e-12f);
  dist[e] = d;
  float x = d * CUTOFF_INV;
  float env = envelope6(x);
  #pragma unroll
  for (int r = 0; r < 6; ++r)
    rbf[e * 6 + r] = env * __sinf((float)(r + 1) * PI_F * x);
}
__global__ void k_triplet(const float* __restrict__ rel, const float* __restrict__ dist,
                          const int* __restrict__ ikj, const int* __restrict__ iji,
                          unsigned short* __restrict__ sbf, int T) {
  int t = blockIdx.x * 256 + threadIdx.x;
  if (t >= T) return;
  int ji = iji[t], kj = ikj[t];
  float ax = rel[ji * 3], ay = rel[ji * 3 + 1], az = rel[ji * 3 + 2];
  float bx = rel[kj * 3], by = rel[kj * 3 + 1], bz = rel[kj * 3 + 2];
  float dot = ax * bx + ay * by + az * bz;
  float cx = ay * bz - az * by, cy = az * bx - ax * bz, cz = ax * by - ay * bx;
  float crs = sqrtf(cx * cx + cy * cy + cz * cz);
  float angle = atan2f(crs, dot);
  float x = dist[kj] * CUTOFF_INV;
  float env = envelope6(x);
  float radial[6];
  #pragma unroll
  for (int r = 0; r < 6; ++r) radial[r] = env * __sinf((float)(r + 1) * PI_F * x);
  unsigned short* row = sbf + (long)t * 64;
  #pragma unroll
  for (int l = 0; l < 7; ++l) {
    float cb = __cosf((float)l * angle);
    #pragma unroll
    for (int r = 0; r < 6; ++r) row[l * 6 + r] = f2bf(cb * radial[r]);
  }
  #pragma unroll
  for (int c = 42; c < 64; ++c) row[c] = 0;
}
// out[e,c] = silu(sum_r rbf[e,r]*W[r,c] + b[c])       (K=6 VALU path)
__global__ void k_rbf_lin_silu(const float* __restrict__ rbf, const float* __restrict__ W,
                               const float* __restrict__ bias, float* __restrict__ out, int E) {
  int idx = blockIdx.x * 256 + threadIdx.x;
  if (idx >= E * 128) return;
  int e = idx >> 7, c = idx & 127;
  float s = bias[c];
  #pragma unroll
  for (int r = 0; r < 6; ++r) s += rbf[e * 6 + r] * W[r * 128 + c];
  out[idx] = silu_f(s);
}
// a[e,c] *= sum_r rbf[e,r]*Wf[r,c]
__global__ void k_rbf_mul(const float* __restrict__ rbf, const float* __restrict__ Wf,
                          float* __restrict__ a, int E) {
  int idx = blockIdx.x * 256 + threadIdx.x;
  if (idx >= E * 128) return;
  int e = idx >> 7, c = idx & 127;
  float s = 0.0f;
  #pragma unroll
  for (int r = 0; r < 6; ++r) s += rbf[e * 6 + r] * Wf[r * 128 + c];
  a[idx] *= s;
}
// atomicAdd(g[ei[e],c], (rbf[e] . W[:,c]) * x[e,c])
__global__ void k_out_scatter(const float* __restrict__ rbf, const float* __restrict__ W,
                              const float* __restrict__ x, const int* __restrict__ ei,
                              float* __restrict__ g, int E) {
  int idx = blockIdx.x * 256 + threadIdx.x;
  if (idx >= E * 128) return;
  int e = idx >> 7, c = idx & 127;
  float s = 0.0f;
  #pragma unroll
  for (int r = 0; r < 6; ++r) s += rbf[e * 6 + r] * W[r * 128 + c];
  atomAddF(&g[(long)ei[e] * 128 + c], s * x[idx]);
}
// x[e,c] = silu(t1[ei[e],c] + t2[ej[e],c] + t3[e,c] + bias[c])
__global__ void k_edge_combine(const float* __restrict__ t1, const float* __restrict__ t2,
                               const float* __restrict__ t3, const float* __restrict__ bias,
                               const int* __restrict__ ei, const int* __restrict__ ej,
                               float* __restrict__ x, int E) {
  int idx = blockIdx.x * 256 + threadIdx.x;
  if (idx >= E * 128) return;
  int e = idx >> 7, c = idx & 127;
  float v = t1[(long)ei[e] * 128 + c] + t2[(long)ej[e] * 128 + c] + t3[idx] + bias[c];
  x[idx] = silu_f(v);
}
__global__ void k_batch_scatter(const float* __restrict__ P, const int* __restrict__ batch,
                                float* __restrict__ out, int N) {
  int idx = blockIdx.x * 256 + threadIdx.x;
  if (idx >= N * 128) return;
  int n = idx >> 7, c = idx & 127;
  atomAddF(&out[(long)batch[n] * 128 + c], P[idx]);
}

// ---------------------------------------------------------------------------
extern "C" void kernel_launch(void* const* d_in, const int* in_sizes, int n_in,
                              void* d_out, int out_size, void* d_ws, size_t ws_size,
                              hipStream_t stream) {
  const int N = in_sizes[0];
  const int E = in_sizes[4];
  const int T = in_sizes[6];

  const int*   z      = (const int*)d_in[0];
  const int*   tag_id = (const int*)d_in[1];
  const float* pos    = (const float*)d_in[2];
  const int*   batch  = (const int*)d_in[3];
  const int*   edge_i = (const int*)d_in[4];
  const int*   edge_j = (const int*)d_in[5];
  const int*   idx_kj = (const int*)d_in[6];
  const int*   idx_ji = (const int*)d_in[7];

  // --- params flattened in JAX pytree order (dict keys sorted) -------------
  int pp = 8;
  auto nf = [&]() -> const float* { return (const float*)d_in[pp++]; };
  const float* embW      = nf();                     // emb [95,128]
  const float* emb_lin_b = nf();                     // emb_lin.b
  const float* emb_lin_w = nf();                     // emb_lin.w [384,128]
  const float* emb_rbf_b = nf();                     // emb_lin_rbf.b
  const float* emb_rbf_w = nf();                     // emb_lin_rbf.w [6,128]
  struct IB {
    const float *lin_b, *lin_w, *down_w, *ji_b, *ji_w, *kj_b, *kj_w;
    const float *rbf1_w, *rbf2_w, *sbf1_w, *sbf2_w, *up_w;
    const float *ra_b[4], *ra_w[4], *rb_b[2], *rb_w[2];
  } ib[4];
  for (int b = 0; b < 4; ++b) {
    ib[b].lin_b  = nf(); ib[b].lin_w  = nf();
    ib[b].down_w = nf();
    ib[b].ji_b   = nf(); ib[b].ji_w   = nf();
    ib[b].kj_b   = nf(); ib[b].kj_w   = nf();
    ib[b].rbf1_w = nf(); ib[b].rbf2_w = nf();
    ib[b].sbf1_w = nf(); ib[b].sbf2_w = nf();
    ib[b].up_w   = nf();
    for (int r = 0; r < 4; ++r) { ib[b].ra_b[r] = nf(); ib[b].ra_w[r] = nf(); }
    for (int r = 0; r < 2; ++r) { ib[b].rb_b[r] = nf(); ib[b].rb_w[r] = nf(); }
  }
  struct OB { const float *out_w, *rbf_w, *up_w, *l_b[3], *l_w[3]; } ob[5];
  for (int b = 0; b < 5; ++b) {
    ob[b].out_w = nf(); ob[b].rbf_w = nf(); ob[b].up_w = nf();
    for (int r = 0; r < 3; ++r) { ob[b].l_b[r] = nf(); ob[b].l_w[r] = nf(); }
  }
  const float* tagW = nf();                          // tag [5,128]

  // --- workspace layout ----------------------------------------------------
  char* ws = (char*)d_ws;
  size_t off = 0;
  auto alloc = [&](size_t bytes) -> void* {
    void* r = ws + off;
    off = (off + bytes + 255) & ~(size_t)255;
    return r;
  };
  const size_t SZ_E128 = (size_t)E * 128 * 4;
  const size_t SZ_N128 = (size_t)N * 128 * 4;
  float* Xb   = (float*)alloc(SZ_E128);
  float* Ab   = (float*)alloc(SZ_E128);
  float* Bb   = (float*)alloc(SZ_E128);
  float* Cb   = (float*)alloc(SZ_E128);
  unsigned short* sbfB = (unsigned short*)alloc((size_t)T * 64 * 2);
  float* relB  = (float*)alloc((size_t)E * 3 * 4);
  float* distB = (float*)alloc((size_t)E * 4);
  float* rbfB  = (float*)alloc((size_t)E * 6 * 4);
  float* hB    = (float*)alloc(SZ_N128);
  float* t1B   = (float*)alloc(SZ_N128);
  float* t2B   = (float*)alloc(SZ_N128);
  float* gN    = (float*)alloc(SZ_N128);
  float* g2    = (float*)alloc(SZ_N128);
  float* Pb    = (float*)alloc(SZ_N128);
  float* rbfF[4];
  for (int b = 0; b < 4; ++b) rbfF[b] = (float*)alloc(6 * 128 * 4);
  float* sbfTmp = (float*)alloc(42 * 128 * 4);

  auto pack = [&](const float* Wsrc, int Kreal, int Kpad) -> const unsigned short* {
    unsigned short* dst = (unsigned short*)alloc((size_t)Kpad * 128 * 2);
    int total = Kpad * 128;
    k_pack_w<<<(total + 255) / 256, 256, 0, stream>>>(Wsrc, dst, Kreal, Kpad);
    return dst;
  };
  auto gemm = [&](const float* X, const unsigned short* W, const float* bias,
                  float* Y, int M, int K, int flags) {
    dim3 grid((M + 63) / 64), blk(128);
    switch (flags) {
      case 0: k_gemm<0><<<grid, blk, 0, stream>>>(X, W, bias, Y, M, K); break;
      case 2: k_gemm<2><<<grid, blk, 0, stream>>>(X, W, bias, Y, M, K); break;
      case 3: k_gemm<3><<<grid, blk, 0, stream>>>(X, W, bias, Y, M, K); break;
      case 4: k_gemm<4><<<grid, blk, 0, stream>>>(X, W, bias, Y, M, K); break;
      case 7: k_gemm<7><<<grid, blk, 0, stream>>>(X, W, bias, Y, M, K); break;
    }
  };
  auto zero = [&](float* p, long long n) {
    k_zero<<<(unsigned)((n + 255) / 256), 256, 0, stream>>>(p, n);
  };
  auto addv = [&](float* y, const float* a, long long n) {
    k_add<<<(unsigned)((n + 255) / 256), 256, 0, stream>>>(y, a, n);
  };
  const unsigned EC = (unsigned)(((long long)E * 128 + 255) / 256);

  // --- pre-pass: fuse & pack weights --------------------------------------
  const unsigned short* P_embA = pack(emb_lin_w,                 128, 128);
  const unsigned short* P_embB = pack(emb_lin_w + 128 * 128,     128, 128);
  const unsigned short* P_embC = pack(emb_lin_w + 256 * 128,     128, 128);
  const unsigned short *P_kj[4], *P_ji[4], *P_down[4], *P_up[4], *P_lin[4];
  const unsigned short *P_rb[4][2], *P_ra[4][4], *P_sbf[4];
  for (int b = 0; b < 4; ++b) {
    P_kj[b]   = pack(ib[b].kj_w, 128, 128);
    P_ji[b]   = pack(ib[b].ji_w, 128, 128);
    P_down[b] = pack(ib[b].down_w, 128, 128);
    P_up[b]   = pack(ib[b].up_w, 128, 128);
    P_lin[b]  = pack(ib[b].lin_w, 128, 128);
    for (int r = 0; r < 2; ++r) P_rb[b][r] = pack(ib[b].rb_w[r], 128, 128);
    for (int r = 0; r < 4; ++r) P_ra[b][r] = pack(ib[b].ra_w[r], 128, 128);
    k_fuse_w<<<(6 * 128 + 255) / 256, 256, 0, stream>>>(ib[b].rbf1_w, ib[b].rbf2_w, rbfF[b], 6);
    k_fuse_w<<<(42 * 128 + 255) / 256, 256, 0, stream>>>(ib[b].sbf1_w, ib[b].sbf2_w, sbfTmp, 42);
    P_sbf[b] = pack(sbfTmp, 42, 64);   // sequential on stream: safe to reuse sbfTmp
  }
  const unsigned short *P_oup[5], *P_ol[5][3], *P_oout[5];
  for (int b = 0; b < 5; ++b) {
    P_oup[b]  = pack(ob[b].up_w, 128, 128);
    for (int r = 0; r < 3; ++r) P_ol[b][r] = pack(ob[b].l_w[r], 128, 128);
    P_oout[b] = pack(ob[b].out_w, 128, 128);
  }

  // --- geometry / features -------------------------------------------------
  zero(Pb, (long long)N * 128);
  zero((float*)d_out, out_size);
  k_embed<<<(N * 128 + 255) / 256, 256, 0, stream>>>(z, tag_id, embW, tagW, hB, N);
  k_edge_geom<<<(E + 255) / 256, 256, 0, stream>>>(pos, edge_i, edge_j, relB, distB, rbfB, E);
  k_triplet<<<(T + 255) / 256, 256, 0, stream>>>(relB, distB, idx_kj, idx_ji, sbfB, T);

  // --- embedding block: x = silu(Wa.h_i + Wb.h_j + Wc.rbf_e + bias) --------
  gemm(hB, P_embA, nullptr, t1B, N, 128, 0);
  gemm(hB, P_embB, nullptr, t2B, N, 128, 0);
  k_rbf_lin_silu<<<EC, 256, 0, stream>>>(rbfB, emb_rbf_w, emb_rbf_b, Ab, E);
  gemm(Ab, P_embC, nullptr, Bb, E, 128, 0);
  k_edge_combine<<<EC, 256, 0, stream>>>(t1B, t2B, Bb, emb_lin_b, edge_i, edge_j, Xb, E);

  // --- output block helper -------------------------------------------------
  auto out_block = [&](int b) {
    zero(gN, (long long)N * 128);
    k_out_scatter<<<EC, 256, 0, stream>>>(rbfB, ob[b].rbf_w, Xb, edge_i, gN, E);
    gemm(gN, P_oup[b],   nullptr,      g2, N, 128, 0);
    gemm(g2, P_ol[b][0], ob[b].l_b[0], gN, N, 128, 3);
    gemm(gN, P_ol[b][1], ob[b].l_b[1], g2, N, 128, 3);
    gemm(g2, P_ol[b][2], ob[b].l_b[2], gN, N, 128, 3);
    gemm(gN, P_oout[b],  nullptr,      Pb, N, 128, 4);   // P += ...
  };

  out_block(0);

  // --- interaction blocks --------------------------------------------------
  for (int b = 0; b < 4; ++b) {
    const long long nE = (long long)E * 128;
    gemm(Xb, P_kj[b], ib[b].kj_b, Ab, E, 128, 3);                   // x_kj
    k_rbf_mul<<<EC, 256, 0, stream>>>(rbfB, rbfF[b], Ab, E);        // * fused rbf
    gemm(Ab, P_down[b], nullptr, Bb, E, 128, 2);                    // x_down
    zero(Cb, nE);
    k_sbf_gemm_scatter<<<T / 64, 128, 0, stream>>>(sbfB, P_sbf[b], Bb,
                                                   idx_kj, idx_ji, Cb, T);
    gemm(Cb, P_up[b], nullptr, Ab, E, 128, 2);                      // x_up
    gemm(Xb, P_ji[b], ib[b].ji_b, Bb, E, 128, 3);                   // x_ji
    addv(Bb, Ab, nE);                                               // h = x_ji + x_up
    gemm(Bb, P_rb[b][0], ib[b].rb_b[0], Ab, E, 128, 3);             // res_before
    gemm(Ab, P_rb[b][1], ib[b].rb_b[1], Bb, E, 128, 7);             // h += silu(..)
    gemm(Bb, P_lin[b], ib[b].lin_b, Ab, E, 128, 3);                 // silu(lin(h))
    addv(Ab, Xb, nE);                                               // + x
    gemm(Ab, P_ra[b][0], ib[b].ra_b[0], Bb, E, 128, 3);             // res_after 1
    gemm(Bb, P_ra[b][1], ib[b].ra_b[1], Ab, E, 128, 7);
    gemm(Ab, P_ra[b][2], ib[b].ra_b[2], Bb, E, 128, 3);             // res_after 2
    gemm(Bb, P_ra[b][3], ib[b].ra_b[3], Ab, E, 128, 7);
    std::swap(Xb, Ab);                                              // new x
    out_block(b + 1);
  }

  // --- graph readout -------------------------------------------------------
  k_batch_scatter<<<(N * 128 + 255) / 256, 256, 0, stream>>>(Pb, batch, (float*)d_out, N);
  (void)n_in; (void)ws_size;
}
